// sLSTM_28449863369520
// MI455X (gfx1250) — compile-verified, matrix operand order checked by reading, not run on hardware
//
#include <hip/hip_runtime.h>

typedef __attribute__((ext_vector_type(16))) __bf16   v16bf;
typedef __attribute__((ext_vector_type(8)))  float    v8f;
typedef __attribute__((ext_vector_type(4)))  unsigned u32x4;

#define T_STEPS 2048
#define BATCH   16
#define IDIM    512
#define HDIM    512
#define WMAT_ELEMS (2048*512)
#define GATE_STRIDE ((size_t)512*512)   // elements between gate blocks in one weight matrix
#define NWG     16

union ABits { u32x4 u[2]; v16bf v; };

// bf16 operand tile load, matching CDNA5 16-bit A/B VGPR layout:
// lanes 0-15 hold K=[0,8)+[16,24); lanes 16-31 hold K=[8,16)+[24,32).
// Two contiguous 16B chunks per lane -> global_load_b128 x2.
__device__ __forceinline__ v16bf ld_bf16_tile(const __bf16* rowptr, int k0, int lane) {
  int koff = ((lane >> 4) & 1) * 8;
  ABits ab;
  ab.u[0] = *(const u32x4*)(rowptr + k0 + koff);
  ab.u[1] = *(const u32x4*)(rowptr + k0 + koff + 16);
  return ab.v;
}

// Same tile shape but sourced from fp32 (layer-0 input_seq), converted on the fly.
__device__ __forceinline__ v16bf ld_f32_tile(const float* rowptr, int k0, int lane) {
  int koff = ((lane >> 4) & 1) * 8;
  const float* p = rowptr + k0 + koff;
  v16bf v;
#pragma unroll
  for (int i = 0; i < 8; ++i) {
    v[i]     = (__bf16)p[i];
    v[i + 8] = (__bf16)p[i + 16];
  }
  return v;
}

// One-shot fp32 -> bf16 weight conversion: [w_ih0, w_hh0, w_ih1, w_hh1]
__global__ void cvt_weights(const float* __restrict__ w0, const float* __restrict__ w1,
                            const float* __restrict__ w2, const float* __restrict__ w3,
                            __bf16* __restrict__ dst) {
  const float* srcs[4] = {w0, w1, w2, w3};
  size_t r = (size_t)blockIdx.x * blockDim.x + threadIdx.x;
  dst[(size_t)blockIdx.y * WMAT_ELEMS + r] = (__bf16)srcs[blockIdx.y][r];
}

__global__ __launch_bounds__(256, 1) void slstm_persist(
    const float* __restrict__ xin,   // [T][B][I] fp32
    const float* __restrict__ b0,
    const float* __restrict__ b1,
    const __bf16* __restrict__ wbf,  // 4 x [2048][512] bf16
    __bf16* __restrict__ hbuf,       // [layer][parity][B][H] bf16
    float* __restrict__ out,
    unsigned* __restrict__ barcnt) {
  __shared__ float red[2][4][4][256];   // [quad][wave-in-quad][gate][tile elem]

  const int tid   = threadIdx.x;
  const int lane  = tid & 31;
  const int wave  = tid >> 5;
  const int quad  = wave >> 2;          // 0,1 : which hidden slice in this WG
  const int q     = wave & 3;           // K-chunk owner within the quad
  const int slice = blockIdx.x * 2 + quad;   // 0..31
  const int u0    = slice * 16;
  const int ln    = lane & 15;          // A: batch row / B: output-unit column
  const int rhalf = (lane >> 4) ? 8 : 0;
  const int kbase = (q & 1) * 256;
  const int ish   = (q >= 2) ? 1 : 0;   // 0: w_ih (x part), 1: w_hh (h part)

  // Single B base pointer; gate blocks addressed by constant 512KB offsets.
  const __bf16* wbase = wbf + (size_t)(u0 + ln) * 512;

  // Hoist per-lane biases (column ln of each gate block, both layers).
  float bias[2][4];
#pragma unroll
  for (int g = 0; g < 4; ++g) {
    bias[0][g] = b0[g * HDIM + u0 + ln];
    bias[1][g] = b1[g * HDIM + u0 + ln];
  }

  // Persistent per-slice LSTM state lives in VGPRs for all 2048 steps.
  v8f c_st[2] = {}, n_st[2] = {};
  unsigned epoch = 0;

  for (int t = 0; t < T_STEPS; ++t) {
    const int wp = t & 1;      // parity written this step
    const int rp = wp ^ 1;     // parity holding h_{t-1}
#pragma unroll
    for (int layer = 0; layer < 2; ++layer) {
      v8f acc[4] = {};
      const __bf16* wr = wbase + (size_t)(layer * 2 + ish) * WMAT_ELEMS;

      const bool srcf32 = (q < 2) && (layer == 0);
      const float* arow_f = xin + (size_t)t * BATCH * IDIM + (size_t)ln * IDIM;
      const __bf16* arow_b =
          (q < 2) ? hbuf + ((size_t)(0 * 2 + wp) * BATCH + ln) * HDIM        // layer1 x = h0 (current)
                  : hbuf + ((size_t)(layer * 2 + rp) * BATCH + ln) * HDIM;   // h_{t-1} of this layer

      if (srcf32) {
#pragma unroll 2
        for (int kk = 0; kk < 256; kk += 32) {
          const int k0 = kbase + kk;
          v16bf a = ld_f32_tile(arow_f, k0, lane);
#pragma unroll
          for (int g = 0; g < 4; ++g) {
            v16bf bm = ld_bf16_tile(wr + (size_t)g * GATE_STRIDE, k0, lane);
            acc[g] = __builtin_amdgcn_wmma_f32_16x16x32_bf16(
                false, a, false, bm, (short)0, acc[g], false, false);
          }
        }
      } else {
#pragma unroll 2
        for (int kk = 0; kk < 256; kk += 32) {
          const int k0 = kbase + kk;
          v16bf a = ld_bf16_tile(arow_b, k0, lane);
#pragma unroll
          for (int g = 0; g < 4; ++g) {
            v16bf bm = ld_bf16_tile(wr + (size_t)g * GATE_STRIDE, k0, lane);
            acc[g] = __builtin_amdgcn_wmma_f32_16x16x32_bf16(
                false, a, false, bm, (short)0, acc[g], false, false);
          }
        }
      }

      // Cross-wave K reduction through LDS (waves 1..3 publish; wave 0 combines).
      if (q != 0) {
#pragma unroll
        for (int g = 0; g < 4; ++g)
#pragma unroll
          for (int r = 0; r < 8; ++r)
            red[quad][q][g][lane * 8 + r] = acc[g][r];
      }
      __syncthreads();

      if (q == 0) {
#pragma unroll
        for (int g = 0; g < 4; ++g)
#pragma unroll
          for (int r = 0; r < 8; ++r)
            acc[g][r] += red[quad][1][g][lane * 8 + r] +
                         red[quad][2][g][lane * 8 + r] +
                         red[quad][3][g][lane * 8 + r];

        __bf16* hout = hbuf + (size_t)(layer * 2 + wp) * BATCH * HDIM;
        const int u = u0 + ln;
#pragma unroll
        for (int r = 0; r < 8; ++r) {
          const int bm = r + rhalf;   // batch row of this C element
          float gi = acc[0][r] + bias[layer][0];
          float gf = acc[1][r] + bias[layer][1];
          float gg = acc[2][r] + bias[layer][2];
          float go = acc[3][r] + bias[layer][3];
          float iv = __expf(gi);
          float fv = 1.0f / (1.0f + __expf(-gf));
          float e2 = __expf(2.0f * gg);
          float gv = (e2 - 1.0f) / (e2 + 1.0f);
          float ov = 1.0f / (1.0f + __expf(-go));
          float nv = fv * n_st[layer][r] + iv;
          float cv = fv * c_st[layer][r] + iv * gv;
          float hv = ov * (cv / (nv + 1e-8f));
          n_st[layer][r] = nv;
          c_st[layer][r] = cv;
          hout[(size_t)bm * HDIM + u] = (__bf16)hv;
          if (layer == 1)
            out[(size_t)t * BATCH * HDIM + (size_t)bm * HDIM + u] = hv;
          if (t == T_STEPS - 1) {
            const size_t base = (size_t)T_STEPS * BATCH * HDIM;
            const size_t lbh  = (size_t)2 * BATCH * HDIM;
            const size_t idx  = (size_t)layer * BATCH * HDIM + (size_t)bm * HDIM + u;
            out[base + idx]           = hv;   // hn
            out[base + lbh + idx]     = cv;   // cn
            out[base + 2 * lbh + idx] = nv;   // nn
          }
        }
      }

      // Grid-wide barrier (monotonic counter, 16 WGs).
      epoch += NWG;
      __threadfence();
      __syncthreads();
      if (tid == 0) {
        __atomic_fetch_add(barcnt, 1u, __ATOMIC_RELEASE);
        while (__atomic_load_n(barcnt, __ATOMIC_ACQUIRE) < epoch)
          __builtin_amdgcn_s_sleep(1);
      }
      __syncthreads();
    }
  }
}

extern "C" void kernel_launch(void* const* d_in, const int* in_sizes, int n_in,
                              void* d_out, int out_size, void* d_ws, size_t ws_size,
                              hipStream_t stream) {
  const float* xin  = (const float*)d_in[0];
  const float* wih0 = (const float*)d_in[1];
  const float* whh0 = (const float*)d_in[2];
  const float* b0   = (const float*)d_in[3];
  const float* wih1 = (const float*)d_in[4];
  const float* whh1 = (const float*)d_in[5];
  const float* b1   = (const float*)d_in[6];

  char* ws = (char*)d_ws;
  unsigned* barcnt = (unsigned*)ws;                    // 256 B (counter + pad)
  __bf16*   hbuf   = (__bf16*)(ws + 256);              // 2*2*16*512 bf16 = 64 KB
  __bf16*   wbf    = (__bf16*)(ws + 256 + 65536);      // 4*2048*512 bf16 = 8 MB

  // Zero barrier counter + h double-buffers (h_{-1} = 0) every launch: replay-safe.
  hipMemsetAsync(ws, 0, 256 + 65536, stream);

  dim3 cg(WMAT_ELEMS / 256, 4);
  cvt_weights<<<cg, 256, 0, stream>>>(wih0, whh0, wih1, whh1, wbf);

  slstm_persist<<<NWG, 256, 0, stream>>>(xin, b0, b1, wbf, hbuf, (float*)d_out, barcnt);
}
